// MotionFormer_5660766896470
// MI455X (gfx1250) — matrix operand order repaired and checked by reading, not compiled, not run on hardware
//
#include <hip/hip_runtime.h>

typedef __bf16 bf16;
typedef bf16  v16bf __attribute__((ext_vector_type(16)));
typedef bf16  v8bf  __attribute__((ext_vector_type(8)));
typedef float v8f   __attribute__((ext_vector_type(8)));

#define BB     16
#define NTOK   1569
#define DIMD   768
#define NHEAD  12
#define FRAMES 8
#define HD     64
#define NSP    196          // (1569-1)/8
#define MROWS  (BB*NTOK)    // 25104

// ---------------------------------------------------------------- convert f32 -> bf16 (no transpose)
__global__ void k_cvt_bf16(const float* __restrict__ s, bf16* __restrict__ d, long long n) {
    long long i  = (long long)blockIdx.x * blockDim.x + threadIdx.x;
    long long st = (long long)gridDim.x * blockDim.x;
    for (; i < n; i += st) d[i] = (bf16)s[i];
}

// ---------------------------------------------------------------- convert+transpose: src f32[K][N] -> dst bf16[N][K]
__global__ __launch_bounds__(256)
void k_cvt_tr_bf16(const float* __restrict__ src, bf16* __restrict__ dst, int K, int N) {
    __shared__ float t[32][33];
    const int nt = blockIdx.x * 32, kt = blockIdx.y * 32;
    const int tx = threadIdx.x, ty = threadIdx.y; // 32 x 8
    #pragma unroll
    for (int i = 0; i < 4; i++) {
        int row = ty + i * 8;
        t[row][tx] = src[(size_t)(kt + row) * N + nt + tx];
    }
    __syncthreads();
    #pragma unroll
    for (int i = 0; i < 4; i++) {
        int row = ty + i * 8;
        dst[(size_t)(nt + row) * K + kt + tx] = (bf16)t[tx][row];
    }
}

// ---------------------------------------------------------------- LayerNorm -> bf16
__global__ __launch_bounds__(256)
void k_ln_bf16(const float* __restrict__ x, const float* __restrict__ g,
               const float* __restrict__ bt, bf16* __restrict__ y) {
    __shared__ float red[256];
    const int row = blockIdx.x, tid = threadIdx.x;
    const float* xr = x + (size_t)row * DIMD;
    float s = 0.f, ss = 0.f;
    for (int c = tid; c < DIMD; c += 256) { float v = xr[c]; s += v; ss += v * v; }
    red[tid] = s; __syncthreads();
    for (int o = 128; o > 0; o >>= 1) { if (tid < o) red[tid] += red[tid + o]; __syncthreads(); }
    float mean = red[0] / (float)DIMD; __syncthreads();
    red[tid] = ss; __syncthreads();
    for (int o = 128; o > 0; o >>= 1) { if (tid < o) red[tid] += red[tid + o]; __syncthreads(); }
    float var = red[0] / (float)DIMD - mean * mean;
    float inv = rsqrtf(var + 1e-6f);
    bf16* yr = y + (size_t)row * DIMD;
    for (int c = tid; c < DIMD; c += 256)
        yr[c] = (bf16)((xr[c] - mean) * inv * g[c] + bt[c]);
}

// ---------------------------------------------------------------- WMMA GEMM, CTA tile 128x128x32
// C[M,N] = A[M,K](bf16, row-major) * WT[N,K](bf16, pre-transposed) + bias
// optional GELU / residual ; out f32 or bf16. 256 threads = 8 waves, wave tile 32x64.
template <bool OUT_BF16, bool DO_GELU, bool DO_RESID>
__global__ __launch_bounds__(256)
void k_gemm(const bf16* __restrict__ A, const bf16* __restrict__ WT,
            const float* __restrict__ bias, const float* __restrict__ resid,
            float* __restrict__ outF, bf16* __restrict__ outB,
            int M, int N, int K) {
    constexpr int ST = 40;                 // element stride (80B rows, 16B aligned groups)
    __shared__ bf16 As[128 * ST];          // 128 rows x 32 k
    __shared__ bf16 Bs[128 * ST];          // 128 cols(n) x 32 k   (WT is n-major, k-contiguous)
    const int tid = threadIdx.x, lane = tid & 31, w = tid >> 5;
    const int wm = w & 3, wn = w >> 2;     // 4 row-strips x 2 col-strips
    const int mBase = blockIdx.y * 128, nBase = blockIdx.x * 128;
    v8f acc[2][4] = {};

    const int r  = tid >> 1;               // 0..127 staging row
    const int ch = (tid & 1) * 16;         // k half
    const int gr = mBase + r;
    const int gn = nBase + r;

    for (int kt = 0; kt < K; kt += 32) {
        { // stage A (row guard on M); coalesced 32B per thread
            bf16* dst = &As[r * ST + ch];
            if (gr < M) {
                const bf16* src = A + (size_t)gr * K + kt + ch;
                #pragma unroll
                for (int i = 0; i < 16; i++) dst[i] = src[i];
            } else {
                #pragma unroll
                for (int i = 0; i < 16; i++) dst[i] = (bf16)0.0f;
            }
        }
        { // stage WT row n (N is a multiple of 128 -> no guard)
            bf16* dst = &Bs[r * ST + ch];
            const bf16* src = WT + (size_t)gn * K + kt + ch;
            #pragma unroll
            for (int i = 0; i < 16; i++) dst[i] = src[i];
        }
        __syncthreads();

        // A fragments for this wave's two 16-row halves (contiguous k = 16*(lane/16) + i, same as B)
        v16bf af[2];
        #pragma unroll
        for (int rh = 0; rh < 2; rh++) {
            int mrow = wm * 32 + rh * 16 + (lane & 15);
            int kh   = (lane >> 4) * 16;
            const bf16* ap = &As[mrow * ST + kh];
            v8bf alo = *(const v8bf*)ap;
            v8bf ahi = *(const v8bf*)(ap + 8);
            #pragma unroll
            for (int i = 0; i < 8; i++) { af[rh][i] = alo[i]; af[rh][i + 8] = ahi[i]; }
        }
        #pragma unroll
        for (int j = 0; j < 4; j++) {
            int ncol = wn * 64 + j * 16 + (lane & 15);
            int kb   = (lane >> 4) * 16;
            const bf16* bp = &Bs[ncol * ST + kb];
            v8bf blo = *(const v8bf*)bp;
            v8bf bhi = *(const v8bf*)(bp + 8);
            v16bf bfr;
            #pragma unroll
            for (int i = 0; i < 8; i++) { bfr[i] = blo[i]; bfr[i + 8] = bhi[i]; }
            acc[0][j] = __builtin_amdgcn_wmma_f32_16x16x32_bf16(
                false, af[0], false, bfr, (short)0, acc[0][j], false, false);
            acc[1][j] = __builtin_amdgcn_wmma_f32_16x16x32_bf16(
                false, af[1], false, bfr, (short)0, acc[1][j], false, false);
        }
        __syncthreads();
    }

    // epilogue: C layout — VGPR r -> row r + 8*(lane/16); col = lane%16
    const int cb = nBase + wn * 64 + (lane & 15);
    #pragma unroll
    for (int rh = 0; rh < 2; rh++) {
        int rbase = mBase + wm * 32 + rh * 16 + 8 * (lane >> 4);
        #pragma unroll
        for (int j = 0; j < 4; j++) {
            int col  = cb + j * 16;
            float bv = bias[col];
            #pragma unroll
            for (int rr = 0; rr < 8; rr++) {
                int row = rbase + rr;
                if (row < M) {
                    float v = acc[rh][j][rr] + bv;
                    if (DO_GELU) {
                        float t = 0.7978845608028654f * (v + 0.044715f * v * v * v);
                        v = 0.5f * v * (1.0f + tanhf(t));
                    }
                    if (DO_RESID) v += resid[(size_t)row * N + col];
                    if (OUT_BF16) outB[(size_t)row * N + col] = (bf16)v;
                    else          outF[(size_t)row * N + col] = v;
                }
            }
        }
    }
}

// ---------------------------------------------------------------- cls-token attention (1 q x 1569 k) per (b,h)
__global__ __launch_bounds__(256)
void k_attn_cls(const float* __restrict__ qkv, float* __restrict__ attn) {
    __shared__ float qs[64], sc[NTOK], red[256], outp[4 * 64];
    const int bid = blockIdx.x, h = bid % NHEAD, b = bid / NHEAD, tid = threadIdx.x;
    if (tid < 64) qs[tid] = qkv[((size_t)(b * NTOK)) * 2304 + h * HD + tid];
    __syncthreads();
    float lmax = -1e30f;
    for (int j = tid; j < NTOK; j += 256) {
        size_t base = ((size_t)(b * NTOK + j)) * 2304 + DIMD + h * HD;
        float s = 0.f;
        for (int d = 0; d < HD; d++) s += qs[d] * qkv[base + d];
        s *= 0.125f; sc[j] = s; lmax = fmaxf(lmax, s);
    }
    red[tid] = lmax; __syncthreads();
    for (int o = 128; o > 0; o >>= 1) { if (tid < o) red[tid] = fmaxf(red[tid], red[tid + o]); __syncthreads(); }
    float m = red[0]; __syncthreads();
    float lsum = 0.f;
    for (int j = tid; j < NTOK; j += 256) { float p = __expf(sc[j] - m); sc[j] = p; lsum += p; }
    red[tid] = lsum; __syncthreads();
    for (int o = 128; o > 0; o >>= 1) { if (tid < o) red[tid] += red[tid + o]; __syncthreads(); }
    float denom = red[0]; __syncthreads();
    int part = tid >> 6, d = tid & 63;
    float acc = 0.f;
    for (int j = part; j < NTOK; j += 4)
        acc += sc[j] * qkv[((size_t)(b * NTOK + j)) * 2304 + 2 * DIMD + h * HD + d];
    outp[part * 64 + d] = acc; __syncthreads();
    if (tid < 64) {
        float o = (outp[tid] + outp[64 + tid] + outp[128 + tid] + outp[192 + tid]) / denom;
        attn[((size_t)(b * NTOK)) * DIMD + h * HD + tid] = o;
    }
}

// ---------------------------------------------------------------- time attention: 8q x 9k per (b,h,n)
__global__ __launch_bounds__(32)
void k_attn_time(const float* __restrict__ qkv, float* __restrict__ attn) {
    __shared__ float qs[8 * 64], ks[9 * 64], vs[9 * 64], scores[72], probs[72];
    const int bid = blockIdx.x;
    const int n = bid % NSP, h = (bid / NSP) % NHEAD, b = bid / (NSP * NHEAD);
    const int lane = threadIdx.x;
    for (int e = lane; e < 8 * 64; e += 32) {
        int qi = e >> 6, d = e & 63;
        int t = 1 + qi * NSP + n;
        qs[e] = qkv[((size_t)(b * NTOK + t)) * 2304 + h * HD + d];
    }
    for (int e = lane; e < 9 * 64; e += 32) {
        int j = e >> 6, d = e & 63;
        int t = (j == 0) ? 0 : 1 + (j - 1) * NSP + n;
        size_t base = ((size_t)(b * NTOK + t)) * 2304 + h * HD + d;
        ks[e] = qkv[base + DIMD];
        vs[e] = qkv[base + 2 * DIMD];
    }
    __syncthreads();
    for (int p = lane; p < 72; p += 32) {
        int qi = p / 9, kj = p % 9;
        float s = 0.f;
        for (int d = 0; d < HD; d++) s += qs[qi * 64 + d] * ks[kj * 64 + d];
        scores[p] = s * 0.125f;
    }
    __syncthreads();
    if (lane < 8) {
        float m = -1e30f;
        for (int j = 0; j < 9; j++) m = fmaxf(m, scores[lane * 9 + j]);
        float sum = 0.f, e9[9];
        for (int j = 0; j < 9; j++) { e9[j] = __expf(scores[lane * 9 + j] - m); sum += e9[j]; }
        for (int j = 0; j < 9; j++) probs[lane * 9 + j] = e9[j] / sum;
    }
    __syncthreads();
    for (int idx = lane; idx < 512; idx += 32) {
        int qi = idx >> 6, d = idx & 63;
        float o = 0.f;
        for (int j = 0; j < 9; j++) o += probs[qi * 9 + j] * vs[j * 64 + d];
        int t = 1 + qi * NSP + n;
        attn[((size_t)(b * NTOK + t)) * DIMD + h * HD + d] = o;
    }
}

// ---------------------------------------------------------------- space attention: 196q x 197k per (b,h,f)
__global__ __launch_bounds__(256)
void k_attn_space(const float* __restrict__ qkv, float* __restrict__ attn) {
    __shared__ bf16  Ks[197 * 64];
    __shared__ bf16  Vs[197 * 64];
    __shared__ float qs[8 * 64];
    __shared__ float probs[8 * 200];
    const int bid = blockIdx.x;
    const int f = bid & 7, h = (bid >> 3) % NHEAD, b = bid / (FRAMES * NHEAD);
    const int tid = threadIdx.x, lane = tid & 31, w = tid >> 5;
    for (int e = tid; e < 197 * 64; e += 256) {
        int j = e >> 6, d = e & 63;
        int t = (j == 0) ? 0 : 1 + f * NSP + (j - 1);
        size_t base = ((size_t)(b * NTOK + t)) * 2304 + h * HD + d;
        Ks[e] = (bf16)qkv[base + DIMD];
        Vs[e] = (bf16)qkv[base + 2 * DIMD];
    }
    __syncthreads();
    for (int qb = 0; qb < NSP; qb += 8) {
        for (int e = tid; e < 8 * 64; e += 256) {
            int qw = e >> 6, d = e & 63;
            int qi = qb + qw;
            float v = 0.f;
            if (qi < NSP) {
                int t = 1 + f * NSP + qi;
                v = qkv[((size_t)(b * NTOK + t)) * 2304 + h * HD + d];
            }
            qs[e] = v;
        }
        __syncthreads();
        int qi = qb + w;
        if (qi < NSP) {
            float sc[7]; int nk = 0;
            float lmax = -1e30f;
            for (int j = lane; j < 197; j += 32) {
                float s = 0.f;
                for (int d = 0; d < HD; d++) s += qs[w * 64 + d] * (float)Ks[j * 64 + d];
                s *= 0.125f;
                sc[nk++] = s;
                lmax = fmaxf(lmax, s);
            }
            for (int o = 16; o > 0; o >>= 1) lmax = fmaxf(lmax, __shfl_xor(lmax, o, 32));
            float lsum = 0.f;
            for (int i = 0; i < nk; i++) { sc[i] = __expf(sc[i] - lmax); lsum += sc[i]; }
            for (int o = 16; o > 0; o >>= 1) lsum += __shfl_xor(lsum, o, 32);
            float rinv = 1.0f / lsum;
            nk = 0;
            for (int j = lane; j < 197; j += 32) probs[w * 200 + j] = sc[nk++] * rinv;
        }
        __syncthreads();
        if (qi < NSP) {
            int t = 1 + f * NSP + qi;
            for (int d = lane; d < HD; d += 32) {
                float o = 0.f;
                for (int j = 0; j < 197; j++) o += probs[w * 200 + j] * (float)Vs[j * 64 + d];
                attn[((size_t)(b * NTOK + t)) * DIMD + h * HD + d] = o;
            }
        }
        __syncthreads();
    }
}

// ---------------------------------------------------------------- launch
extern "C" void kernel_launch(void* const* d_in, const int* in_sizes, int n_in,
                              void* d_out, int out_size, void* d_ws, size_t ws_size,
                              hipStream_t stream) {
    const float* x    = (const float*)d_in[0];
    const float* n1w  = (const float*)d_in[1];
    const float* n1b  = (const float*)d_in[2];
    const float* n2w  = (const float*)d_in[3];
    const float* n2b  = (const float*)d_in[4];
    const float* n3w  = (const float*)d_in[5];
    const float* n3b  = (const float*)d_in[6];
    const float* tqkv_w = (const float*)d_in[7];
    const float* tqkv_b = (const float*)d_in[8];
    const float* tprj_w = (const float*)d_in[9];
    const float* tprj_b = (const float*)d_in[10];
    const float* sqkv_w = (const float*)d_in[11];
    const float* sqkv_b = (const float*)d_in[12];
    const float* sprj_w = (const float*)d_in[13];
    const float* sprj_b = (const float*)d_in[14];
    const float* m1w = (const float*)d_in[15];
    const float* m1b = (const float*)d_in[16];
    const float* m2w = (const float*)d_in[17];
    const float* m2b = (const float*)d_in[18];
    float* out = (float*)d_out;

    char* ws = (char*)d_ws;
    auto aln = [](size_t v) { return (v + 255) & ~(size_t)255; };
    size_t off = 0;
    bf16* w_tqkv = (bf16*)(ws + off); off = aln(off + (size_t)DIMD * 3 * DIMD * 2);   // [2304][768]
    bf16* w_tprj = (bf16*)(ws + off); off = aln(off + (size_t)DIMD * DIMD * 2);       // [768][768]
    bf16* w_sqkv = (bf16*)(ws + off); off = aln(off + (size_t)DIMD * 3 * DIMD * 2);
    bf16* w_sprj = (bf16*)(ws + off); off = aln(off + (size_t)DIMD * DIMD * 2);
    bf16* w_m1   = (bf16*)(ws + off); off = aln(off + (size_t)DIMD * 4 * DIMD * 2);   // [3072][768]
    bf16* w_m2   = (bf16*)(ws + off); off = aln(off + (size_t)4 * DIMD * DIMD * 2);   // [768][3072]
    float* qkvbuf  = (float*)(ws + off); off = aln(off + (size_t)MROWS * 2304 * 4);   // reused as bf16 MLP act
    bf16*  actbuf  = (bf16*)(ws + off);  off = aln(off + (size_t)MROWS * DIMD * 2);
    float* attnbuf = (float*)(ws + off); off = aln(off + (size_t)MROWS * DIMD * 4);
    (void)ws_size; (void)n_in; (void)in_sizes; (void)out_size;

    const int MT = (MROWS + 127) / 128; // 197 row tiles
    dim3 trb(32, 8);
    // weight conversion + transpose (stateless, every call): src f32[K][N] -> dst bf16[N][K]
    k_cvt_tr_bf16<<<dim3(2304 / 32,  768 / 32), trb, 0, stream>>>(tqkv_w, w_tqkv, DIMD, 3 * DIMD);
    k_cvt_tr_bf16<<<dim3( 768 / 32,  768 / 32), trb, 0, stream>>>(tprj_w, w_tprj, DIMD, DIMD);
    k_cvt_tr_bf16<<<dim3(2304 / 32,  768 / 32), trb, 0, stream>>>(sqkv_w, w_sqkv, DIMD, 3 * DIMD);
    k_cvt_tr_bf16<<<dim3( 768 / 32,  768 / 32), trb, 0, stream>>>(sprj_w, w_sprj, DIMD, DIMD);
    k_cvt_tr_bf16<<<dim3(3072 / 32,  768 / 32), trb, 0, stream>>>(m1w, w_m1, DIMD, 4 * DIMD);
    k_cvt_tr_bf16<<<dim3( 768 / 32, 3072 / 32), trb, 0, stream>>>(m2w, w_m2, 4 * DIMD, DIMD);

    // ---- TIME attention branch (uses norm3) ----
    k_ln_bf16<<<MROWS, 256, 0, stream>>>(x, n3w, n3b, actbuf);
    k_gemm<false, false, false><<<dim3(18, MT), 256, 0, stream>>>(
        actbuf, w_tqkv, tqkv_b, nullptr, qkvbuf, nullptr, MROWS, 2304, DIMD);
    k_attn_cls<<<BB * NHEAD, 256, 0, stream>>>(qkvbuf, attnbuf);
    k_attn_time<<<BB * NHEAD * NSP, 32, 0, stream>>>(qkvbuf, attnbuf);
    k_cvt_bf16<<<2048, 256, 0, stream>>>(attnbuf, actbuf, (long long)MROWS * DIMD);
    k_gemm<false, false, true><<<dim3(6, MT), 256, 0, stream>>>(
        actbuf, w_tprj, tprj_b, x, out, nullptr, MROWS, DIMD, DIMD);     // out = x + t

    // ---- SPACE attention branch (uses norm1) ----
    k_ln_bf16<<<MROWS, 256, 0, stream>>>(out, n1w, n1b, actbuf);
    k_gemm<false, false, false><<<dim3(18, MT), 256, 0, stream>>>(
        actbuf, w_sqkv, sqkv_b, nullptr, qkvbuf, nullptr, MROWS, 2304, DIMD);
    k_attn_cls<<<BB * NHEAD, 256, 0, stream>>>(qkvbuf, attnbuf);
    k_attn_space<<<BB * NHEAD * FRAMES, 256, 0, stream>>>(qkvbuf, attnbuf);
    k_cvt_bf16<<<2048, 256, 0, stream>>>(attnbuf, actbuf, (long long)MROWS * DIMD);
    k_gemm<false, false, true><<<dim3(6, MT), 256, 0, stream>>>(
        actbuf, w_sprj, sprj_b, out, out, nullptr, MROWS, DIMD, DIMD);   // out = x1 + s

    // ---- MLP (uses norm2) ----
    k_ln_bf16<<<MROWS, 256, 0, stream>>>(out, n2w, n2b, actbuf);
    k_gemm<true, true, false><<<dim3(24, MT), 256, 0, stream>>>(
        actbuf, w_m1, m1b, nullptr, nullptr, (bf16*)qkvbuf, MROWS, 3072, DIMD); // GELU -> bf16
    k_gemm<false, false, true><<<dim3(6, MT), 256, 0, stream>>>(
        (bf16*)qkvbuf, w_m2, m2b, out, out, nullptr, MROWS, DIMD, 3072); // out = x2 + mlp
}
